// GAT_13589276525103
// MI455X (gfx1250) — matrix operand order repaired and checked by reading, not compiled, not run on hardware
//
#include <hip/hip_runtime.h>
#include <hip/hip_bf16.h>

typedef __attribute__((ext_vector_type(16))) _Float16 v16h;
typedef __attribute__((ext_vector_type(8)))  _Float16 v8h;
typedef __attribute__((ext_vector_type(2)))  _Float16 v2h;
typedef __attribute__((ext_vector_type(8)))  float    v8f;
typedef __attribute__((ext_vector_type(4)))  float    v4f;

// ---------------------------------------------------------------------------
// One-time prep: W fp32 [k][n] -> half, transposed to Wh[n*128 + k].
// In this layout each WMMA B-fragment is 16 contiguous halves per lane.
// ---------------------------------------------------------------------------
__global__ void convert_w_t(const float* __restrict__ W, _Float16* __restrict__ Wh)
{
    unsigned i = blockIdx.x * blockDim.x + threadIdx.x;   // 16384 threads
    if (i < 128u * 128u) {
        unsigned k = i >> 7, n = i & 127;                  // coalesced read
        Wh[n * 128 + k] = (_Float16)W[i];
    }
}

// ---------------------------------------------------------------------------
// Fused projection GEMM + attention projections (+ optional input ReLU).
// H[n,:] = X[n,:] @ W  (128x128, fp16 WMMA, fp32 accum)
// el[n,h] = dot(H[n, h*32:(h+1)*32], al[h,:]);  er likewise.
// Block: 256 threads = 8 waves; 64-row slab = 4 M-tiles; wave w owns cols
// [16w, 16w+16).  H/el/er padded to a multiple of 64 rows -> no store guards.
// W staged via async global->LDS (ASYNCcnt); al/ar staged to LDS once.
// ---------------------------------------------------------------------------
__global__ __launch_bounds__(256)
void gat_gemm_proj(const float* __restrict__ X, const _Float16* __restrict__ Wh,
                   const float* __restrict__ al, const float* __restrict__ ar,
                   float* __restrict__ H, float* __restrict__ el,
                   float* __restrict__ er, int N, int doRelu)
{
    __shared__ _Float16 sW[128 * 136];   // [col][k], padded pitch (272 B, 16-B mult)
    __shared__ _Float16 sX[64 * 136];    // [row][k]
    __shared__ float    sH[16 * 132];    // one 16-row M-tile of H
    __shared__ float    sAl[128];        // attn_l, staged once
    __shared__ float    sAr[128];        // attn_r, staged once

    const int t     = threadIdx.x;
    const int wave  = t >> 5;
    const int lane  = t & 31;
    const int lrow  = lane & 15;
    const int hi    = lane >> 4;         // lane half: 0 or 1
    const int node0 = blockIdx.x * 64;
    const int col   = wave * 16 + lrow;

    // Stage W (already half + transposed): async global->LDS 16-byte copies.
    // Generic LDS pointer low 32 bits == LDS byte offset (aperture layout).
    #pragma unroll
    for (int i = 0; i < 8; ++i) {
        int idx = t + i * 256;                 // 2048 16-byte chunks
        int base = idx * 8;
        int n = base >> 7, k = base & 127;
        unsigned lds_off = (unsigned)(uintptr_t)&sW[n * 136 + k];
        const _Float16* gp = &Wh[base];
        asm volatile("global_load_async_to_lds_b128 %0, %1, off"
                     :: "v"(lds_off), "v"(gp) : "memory");
    }
    // Stage attention vectors (256 floats, 1 per thread).
    if (t < 128)      sAl[t] = al[t];
    else              sAr[t - 128] = ar[t - 128];

    // Stage X slab (fp32 -> fp16), packed pair stores; optional fused ReLU.
    for (int p = t; p < 64 * 64; p += 256) {   // 4096 half-pairs
        int r = p >> 6, kp = (p & 63) * 2;
        int node = node0 + r;
        v2h pk;
        if (node < N) {                        // inputs not padded: keep load guard
            const float* xp = &X[(size_t)node * 128 + kp];
            float f0 = xp[0], f1 = xp[1];
            if (doRelu) { f0 = fmaxf(f0, 0.f); f1 = fmaxf(f1, 0.f); }
            pk[0] = (_Float16)f0;
            pk[1] = (_Float16)f1;
        } else {
            pk[0] = (_Float16)0.0f; pk[1] = (_Float16)0.0f;
        }
        *(v2h*)&sX[r * 136 + kp] = pk;
    }
    asm volatile("s_wait_asynccnt 0x0" ::: "memory");
    __syncthreads();

    // B fragments depend only on (col, kb): load once, reuse for all 4 M-tiles.
    v16h bfrag[4];
    #pragma unroll
    for (int kb = 0; kb < 4; ++kb)
        bfrag[kb] = *(const v16h*)&sW[col * 136 + kb * 32 + hi * 16];

    for (int mt = 0; mt < 4; ++mt) {
        const int row16 = mt * 16;
        v8f c = {};
        #pragma unroll
        for (int kb = 0; kb < 4; ++kb) {
            // A fragment (16x32 f16), contiguous layout: lane row = lrow;
            // halves 0..15 -> K = kb*32 + hi*16 + i
            const _Float16* arow = &sX[(row16 + lrow) * 136 + kb * 32];
            v8h a_lo = *(const v8h*)&arow[hi * 16];
            v8h a_hi = *(const v8h*)&arow[hi * 16 + 8];
            v16h a = __builtin_shufflevector(a_lo, a_hi,
                                             0, 1, 2, 3, 4, 5, 6, 7,
                                             8, 9, 10, 11, 12, 13, 14, 15);
            c = __builtin_amdgcn_wmma_f32_16x16x32_f16(false, a, false, bfrag[kb],
                                                       (short)0, c, false, false);
        }

        // C/D layout: vgpr r -> row = r + hi*8, col = 16*wave + lrow.
        // H rows padded to multiple of 64 -> unconditional stores.
        float* hp = &H[(size_t)(node0 + row16 + hi * 8) * 128 + col];
        #pragma unroll
        for (int r = 0; r < 8; ++r) {
            sH[(r + hi * 8) * 132 + col] = c[r];
            hp[(size_t)r * 128] = c[r];            // immediate-offset stores
        }
        __syncthreads();

        // el/er for this M-tile: 64 threads, one (row, head) each; LDS operands.
        if (t < 64) {
            int row = t >> 2, head = t & 3;
            int node = node0 + row16 + row;        // < padded rows: no guard
            float dl = 0.f, dr = 0.f;
            #pragma unroll
            for (int j4 = 0; j4 < 8; ++j4) {
                v4f hv = *(const v4f*)&sH[row * 132 + head * 32 + j4 * 4];
                v4f av = *(const v4f*)&sAl[head * 32 + j4 * 4];
                v4f bv = *(const v4f*)&sAr[head * 32 + j4 * 4];
                #pragma unroll
                for (int q = 0; q < 4; ++q) {
                    dl += hv[q] * av[q];
                    dr += hv[q] * bv[q];
                }
            }
            el[(size_t)node * 4 + head] = dl;
            er[(size_t)node * 4 + head] = dr;
        }
        __syncthreads();
    }
}

// ---------------------------------------------------------------------------
// Edge-softmax phase kernels
// ---------------------------------------------------------------------------
__global__ void init_ms(float* __restrict__ m, float* __restrict__ s, int n4)
{
    unsigned i = blockIdx.x * blockDim.x + threadIdx.x;   // over N (vec4)
    if (i < (unsigned)n4) {
        v4f mi = { -__builtin_inff(), -__builtin_inff(),
                   -__builtin_inff(), -__builtin_inff() };
        v4f zi = { 0.f, 0.f, 0.f, 0.f };
        *(v4f*)&m[(size_t)i * 4] = mi;
        *(v4f*)&s[(size_t)i * 4] = zi;
    }
}

__device__ __forceinline__ float leaky(float v) { return v > 0.f ? v : 0.2f * v; }

__global__ void edge_max(const int* __restrict__ src, const int* __restrict__ dst,
                         const float* __restrict__ el, const float* __restrict__ er,
                         float* __restrict__ m, int E)
{
    unsigned e = blockIdx.x * blockDim.x + threadIdx.x;
    if (e >= (unsigned)E) return;
    int sn = src[e], dn = dst[e];
    v4f elv = *(const v4f*)&el[(size_t)sn * 4];
    v4f erv = *(const v4f*)&er[(size_t)dn * 4];
    #pragma unroll
    for (int h = 0; h < 4; ++h)
        atomicMax(&m[(size_t)dn * 4 + h], leaky(elv[h] + erv[h]));
}

__global__ void edge_sum(const int* __restrict__ src, const int* __restrict__ dst,
                         const float* __restrict__ el, const float* __restrict__ er,
                         const float* __restrict__ m, float* __restrict__ s, int E)
{
    unsigned e = blockIdx.x * blockDim.x + threadIdx.x;
    if (e >= (unsigned)E) return;
    int sn = src[e], dn = dst[e];
    v4f elv = *(const v4f*)&el[(size_t)sn * 4];
    v4f erv = *(const v4f*)&er[(size_t)dn * 4];
    v4f mv  = *(const v4f*)&m[(size_t)dn * 4];
    #pragma unroll
    for (int h = 0; h < 4; ++h) {
        float v = leaky(elv[h] + erv[h]);
        unsafeAtomicAdd(&s[(size_t)dn * 4 + h], __expf(v - mv[h]));
    }
}

__global__ void init_bias128(float* __restrict__ o, const float* __restrict__ b, unsigned n4)
{
    unsigned i = blockIdx.x * blockDim.x + threadIdx.x;   // over N*128/4
    if (i < n4) *(v4f*)&o[(size_t)i * 4] = *(const v4f*)&b[(i & 31) * 4];
}

__global__ void init_out_meanbias(float* __restrict__ o, const float* __restrict__ b2, unsigned n4)
{
    unsigned i = blockIdx.x * blockDim.x + threadIdx.x;   // over N*32/4
    if (i < n4) {
        int oo = (i & 7) * 4;                             // 0..28, component base
        v4f r;
        #pragma unroll
        for (int j = 0; j < 4; ++j)
            r[j] = 0.25f * (b2[oo + j] + b2[32 + oo + j] + b2[64 + oo + j] + b2[96 + oo + j]);
        *(v4f*)&o[(size_t)i * 4] = r;
    }
}

// One wave per edge: lanes stream the 128-float message (4 heads x 32 lanes).
// Edge index is wave-uniform -> scalarize the per-edge softmax operands.
__global__ void agg128(const int* __restrict__ src, const int* __restrict__ dst,
                       const float* __restrict__ el, const float* __restrict__ er,
                       const float* __restrict__ m, const float* __restrict__ s,
                       const float* __restrict__ H, float* __restrict__ out, int E)
{
    unsigned gid = blockIdx.x * blockDim.x + threadIdx.x;
    unsigned e = __builtin_amdgcn_readfirstlane(gid >> 5);
    if (e >= (unsigned)E) return;
    int lane = gid & 31;
    int sn = src[e], dn = dst[e];
    v4f elv = *(const v4f*)&el[(size_t)sn * 4];
    v4f erv = *(const v4f*)&er[(size_t)dn * 4];
    v4f mv  = *(const v4f*)&m[(size_t)dn * 4];
    v4f sv  = *(const v4f*)&s[(size_t)dn * 4];
    #pragma unroll
    for (int h = 0; h < 4; ++h) {
        float v = leaky(elv[h] + erv[h]);
        float a = __expf(v - mv[h]) / sv[h];
        int colh = h * 32 + lane;
        unsafeAtomicAdd(&out[(size_t)dn * 128 + colh],
                        a * H[(size_t)sn * 128 + colh]);
    }
}

// Layer-2 aggregate with head-mean folded in: one atomic per edge-lane.
__global__ void agg32_mean(const int* __restrict__ src, const int* __restrict__ dst,
                           const float* __restrict__ el, const float* __restrict__ er,
                           const float* __restrict__ m, const float* __restrict__ s,
                           const float* __restrict__ H, float* __restrict__ out, int E)
{
    unsigned gid = blockIdx.x * blockDim.x + threadIdx.x;
    unsigned e = __builtin_amdgcn_readfirstlane(gid >> 5);
    if (e >= (unsigned)E) return;
    int lane = gid & 31;
    int sn = src[e], dn = dst[e];
    v4f elv = *(const v4f*)&el[(size_t)sn * 4];
    v4f erv = *(const v4f*)&er[(size_t)dn * 4];
    v4f mv  = *(const v4f*)&m[(size_t)dn * 4];
    v4f sv  = *(const v4f*)&s[(size_t)dn * 4];
    float acc = 0.f;
    #pragma unroll
    for (int h = 0; h < 4; ++h) {
        float v = leaky(elv[h] + erv[h]);
        float a = __expf(v - mv[h]) / sv[h];
        acc += 0.25f * a * H[(size_t)sn * 128 + h * 32 + lane];
    }
    unsafeAtomicAdd(&out[(size_t)dn * 32 + lane], acc);
}

// ---------------------------------------------------------------------------
extern "C" void kernel_launch(void* const* d_in, const int* in_sizes, int n_in,
                              void* d_out, int out_size, void* d_ws, size_t ws_size,
                              hipStream_t stream)
{
    const float* feat = (const float*)d_in[0];
    const int*   src  = (const int*)d_in[1];
    const int*   dst  = (const int*)d_in[2];
    const float* W1   = (const float*)d_in[3];
    const float* al1  = (const float*)d_in[4];
    const float* ar1  = (const float*)d_in[5];
    const float* b1   = (const float*)d_in[6];
    const float* W2   = (const float*)d_in[7];
    const float* al2  = (const float*)d_in[8];
    const float* ar2  = (const float*)d_in[9];
    const float* b2   = (const float*)d_in[10];

    const int N  = in_sizes[0] / 128;
    const int E  = in_sizes[1];
    const int Nr = (N + 63) & ~63;        // rows padded to GEMM slab size

    float* ws = (float*)d_ws;
    float* A  = ws;                        // [Nr,128]  H (layer 1 then layer 2)
    float* B  = A  + (size_t)Nr * 128;     // [Nr,128]  aggregate / X2
    float* el = B  + (size_t)Nr * 128;     // [Nr,4]
    float* er = el + (size_t)Nr * 4;       // [Nr,4]
    float* m  = er + (size_t)Nr * 4;       // [Nr,4]
    float* s  = m  + (size_t)Nr * 4;       // [Nr,4]
    _Float16* Wh = (_Float16*)(s + (size_t)Nr * 4);  // [128,128] half, transposed
    float* out = (float*)d_out;            // [N,32]

    const dim3 blk(256);
    const int gGemm = Nr / 64;
    const int gN    = (N + 255) / 256;                       // init_ms: 1 thread/node
    const int gE    = (E + 255) / 256;
    const int gE32  = (int)(((long long)E * 32 + 255) / 256);
    const unsigned n128v = (unsigned)N * 32u;                // N*128/4 vec4 elems
    const int gN128v = (int)((n128v + 255u) / 256u);
    const unsigned n32v = (unsigned)N * 8u;                  // N*32/4 vec4 elems
    const int gN32v  = (int)((n32v + 255u) / 256u);

    // ---------------- layer 1 ----------------
    convert_w_t<<<64, blk, 0, stream>>>(W1, Wh);
    gat_gemm_proj<<<gGemm, blk, 0, stream>>>(feat, Wh, al1, ar1, A, el, er, N, 0);
    init_ms<<<gN, blk, 0, stream>>>(m, s, N);
    edge_max<<<gE, blk, 0, stream>>>(src, dst, el, er, m, E);
    edge_sum<<<gE, blk, 0, stream>>>(src, dst, el, er, m, s, E);
    init_bias128<<<gN128v, blk, 0, stream>>>(B, b1, n128v);
    agg128<<<gE32, blk, 0, stream>>>(src, dst, el, er, m, s, A, B, E);

    // ---------------- layer 2 (ReLU fused into GEMM input staging) ----------
    convert_w_t<<<64, blk, 0, stream>>>(W2, Wh);
    gat_gemm_proj<<<gGemm, blk, 0, stream>>>(B, Wh, al2, ar2, A, el, er, N, 1);
    init_ms<<<gN, blk, 0, stream>>>(m, s, N);
    edge_max<<<gE, blk, 0, stream>>>(src, dst, el, er, m, E);
    edge_sum<<<gE, blk, 0, stream>>>(src, dst, el, er, m, s, E);
    init_out_meanbias<<<gN32v, blk, 0, stream>>>(out, b2, n32v);
    agg32_mean<<<gE32, blk, 0, stream>>>(src, dst, el, er, m, s, A, out, E);
}
